// AffEncoder_53068615910133
// MI455X (gfx1250) — matrix operand order repaired and checked by reading, compiled, not run on hardware
//
#include <hip/hip_runtime.h>

// ---------------------------------------------------------------------------
// MI455X (gfx1250) ST-GCN encoder — bf16 WMMA pipeline, LDS-staged weights.
//   GEMM1: 272 x 576p x 32768   (conv1 + A1 einsum + BN1)
//   GEMM2:  96 x 2592p x 32768  (body-part gather + conv2 + A2 einsum + BN2)
//   GEMM3:  16 x  480  x 32768  (conv1d k5 + BN3 + leaky)
//   GEMM4:  64 x   64p x 32768  (conv1d k3 + BN4 + leaky)
// A fragments pre-swizzled to WMMA lane layout, staged to LDS per workgroup
// via async global->LDS copies (ASYNCcnt) so all 8 waves share one L2 read.
// ---------------------------------------------------------------------------

typedef __attribute__((ext_vector_type(16))) __bf16 v16bf;
typedef __attribute__((ext_vector_type(8)))  __bf16 v8bf;
typedef __attribute__((ext_vector_type(8)))  float  v8f;
typedef __attribute__((ext_vector_type(4)))  int    vi4;

#define DEV static __device__ __forceinline__

#if __has_builtin(__builtin_amdgcn_global_load_async_to_lds_b128) && \
    __has_builtin(__builtin_amdgcn_s_wait_asynccnt)
#define USE_ASYNC_LDS 1
#endif

namespace {

constexpr int T = 2048;
constexpr int NSEQ = 16;
constexpr int CTILES = NSEQ * T / 16;  // 2048 column tiles of 16 t-values

constexpr int G1_RT = 17, G1_CH = 18, F1_STRIDE = 288;  // f1bn pad 272->288
constexpr int G2_RT = 6,  G2_CH = 81, F2_STRIDE = 96;
constexpr int G3_CH = 15;
constexpr int G4_RT = 4,  G4_CH = 2;

// body-part map: joint -> (part u, index-in-part jl, part length L)
__constant__ int c_bp_u[17]  = {1,2,3,1,1,4,4,5,5,2,2,3,3,0,0,0,0};
__constant__ int c_bp_jl[17] = {0,0,0,1,2,0,1,0,1,1,2,1,2,0,1,2,3};
__constant__ int c_bp_L[17]  = {3,3,3,3,3,2,2,2,2,3,3,3,3,4,4,4,4};

// 16-bit A-matrix 16x32 fragment K index (contiguous: k = 16*half + i, wave32)
DEV int a_klocal(int i, int half) {
  return i + half * 16;
}

DEV v8f zero8f() {
  v8f z;
#pragma unroll
  for (int j = 0; j < 8; ++j) z[j] = 0.0f;
  return z;
}

DEV v16bf zero16bf() {
  v16bf z;
#pragma unroll
  for (int j = 0; j < 16; ++j) z[j] = (__bf16)0.0f;
  return z;
}

DEV v8f wmma_bf16(v16bf a, v16bf b, v8f c) {
  return __builtin_amdgcn_wmma_f32_16x16x32_bf16(false, a, false, b, (short)0, c,
                                                 false, false);
}

DEV float leaky(float x) { return x > 0.0f ? x : 0.01f * x; }

// 16B global->LDS copy: async (ASYNCcnt) when available, else through VGPRs
DEV void copy16_lds(uint4* lds_dst, const uint4* gsrc) {
#ifdef USE_ASYNC_LDS
  __builtin_amdgcn_global_load_async_to_lds_b128(
      (__attribute__((address_space(1))) vi4*)(uint4*)gsrc,
      (__attribute__((address_space(3))) vi4*)lds_dst, 0, 0);
#else
  *lds_dst = *gsrc;
#endif
}

DEV void wait_async_all() {
#ifdef USE_ASYNC_LDS
  __builtin_amdgcn_s_wait_asynccnt(0);
#endif
}

// ---------------------------------------------------------------------------
// Weight/activation prep kernels (tiny)
// ---------------------------------------------------------------------------

// poses (n,t,51) f32 -> bf16 [n*t][64] zero-padded (aligned B-fragment rows)
__global__ void prep_poses(const float* __restrict__ poses,
                           __bf16* __restrict__ posesb) {
  int tid = blockIdx.x * blockDim.x + threadIdx.x;
  if (tid >= NSEQ * T * 64) return;
  int nt = tid >> 6, ch = tid & 63;
  posesb[tid] = (ch < 51) ? (__bf16)poses[(size_t)nt * 51 + ch] : (__bf16)0.0f;
}

__global__ void prep_w1(const float* __restrict__ w1, const float* __restrict__ A1,
                        __bf16* __restrict__ A1f) {
  int tid = blockIdx.x * blockDim.x + threadIdx.x;
  if (tid >= G1_RT * G1_CH * 512) return;
  int i = tid & 15, lane = (tid >> 4) & 31;
  int kk = (tid >> 9) % G1_CH, rt = tid / (G1_CH * 512);
  int m = lane & 15, half = lane >> 4;
  int r = rt * 16 + m;                  // r = joint*16 + c
  int c = r & 15, w = r >> 4;
  int k = kk * 32 + a_klocal(i, half);
  int kt = k / 64, uc = k % 64;
  float val = 0.0f;
  if (uc < 51) {
    int u = uc / 3, ci = uc % 3;
    int vlo = u - 2 > 0 ? u - 2 : 0;
    int vhi = u + 2 < 16 ? u + 2 : 16;
    for (int kq = 0; kq < 5; ++kq)
      for (int v = vlo; v <= vhi; ++v) {
        int kv = u - v + 2;
        val += w1[(kq * 16 + c) * 135 + ci * 45 + kt * 5 + kv] *
               A1[(kq * 17 + v) * 17 + w];
      }
  }
  A1f[tid] = (__bf16)val;
}

__global__ void prep_w2(const float* __restrict__ w2, const float* __restrict__ A2,
                        __bf16* __restrict__ A2f) {
  int tid = blockIdx.x * blockDim.x + threadIdx.x;
  if (tid >= G2_RT * G2_CH * 512) return;
  int i = tid & 15, lane = (tid >> 4) & 31;
  int kk = (tid >> 9) % G2_CH, rt = tid / (G2_CH * 512);
  int m = lane & 15, half = lane >> 4;
  int r = rt * 16 + m;                  // r = c*6 + w
  int c = r / 6, w = r % 6;
  int kt = kk / 9;
  int kloc = (kk % 9) * 32 + a_klocal(i, half);
  float val = 0.0f;
  if (kloc < 272) {                     // channel = joint*16 + c2
    int j = kloc >> 4, c2 = kloc & 15;
    int u = c_bp_u[j], jl = c_bp_jl[j], L = c_bp_L[j];
    int ch = c2 * L + jl;
    int vlo = u - 1 > 0 ? u - 1 : 0;
    int vhi = u + 1 < 5 ? u + 1 : 5;
    for (int kq = 0; kq < 5; ++kq)
      for (int v = vlo; v <= vhi; ++v) {
        int kv = u - v + 1;
        val += w2[(kq * 16 + c) * 1728 + ch * 27 + kt * 3 + kv] *
               A2[(kq * 6 + v) * 6 + w];
      }
  }
  A2f[tid] = (__bf16)val;
}

__global__ void prep_w3(const float* __restrict__ w3, __bf16* __restrict__ A3f) {
  int tid = blockIdx.x * blockDim.x + threadIdx.x;
  if (tid >= G3_CH * 512) return;
  int i = tid & 15, lane = (tid >> 4) & 31, kk = tid >> 9;
  int m = lane & 15, half = lane >> 4;
  int kt = kk / 3;
  int ch = (kk % 3) * 32 + a_klocal(i, half);
  A3f[tid] = (__bf16)w3[m * 480 + ch * 5 + kt];
}

__global__ void prep_w4(const float* __restrict__ w4, __bf16* __restrict__ A4f) {
  int tid = blockIdx.x * blockDim.x + threadIdx.x;
  if (tid >= G4_RT * G4_CH * 512) return;
  int i = tid & 15, lane = (tid >> 4) & 31;
  int kk = (tid >> 9) & 1, rt = tid >> 10;
  int m = lane & 15, half = lane >> 4;
  int r = rt * 16 + m;
  int k = kk * 32 + a_klocal(i, half);
  float val = (k < 48) ? w4[r * 48 + (k & 15) * 3 + (k >> 4)] : 0.0f;
  A4f[tid] = (__bf16)val;
}

__global__ void prep_aff(const float* b1, const float* A1, const float* g1,
                         const float* be1, const float* m1, const float* v1,
                         const float* b2, const float* A2, const float* g2,
                         const float* be2, const float* m2, const float* v2,
                         const float* b3, const float* g3, const float* be3,
                         const float* m3, const float* v3,
                         const float* b4, const float* g4, const float* be4,
                         const float* m4, const float* v4,
                         float2* aff1, float2* aff2, float2* aff3, float2* aff4) {
  int tid = blockIdx.x * blockDim.x + threadIdx.x;
  const float eps = 1e-5f;
  if (tid < 272) {
    int r = tid, c = r & 15, w = r >> 4, ch = c * 17 + w;
    float beff = 0.0f;
    for (int kq = 0; kq < 5; ++kq) {
      float s = 0.0f;
      for (int v = 0; v < 17; ++v) s += A1[(kq * 17 + v) * 17 + w];
      beff += b1[kq * 16 + c] * s;
    }
    float sc = g1[ch] * rsqrtf(v1[ch] + eps);
    aff1[r] = make_float2(sc, (beff - m1[ch]) * sc + be1[ch]);
  } else if (tid < 368) {
    int r = tid - 272, c = r / 6, w = r % 6, ch = r;
    float beff = 0.0f;
    for (int kq = 0; kq < 5; ++kq) {
      float s = 0.0f;
      for (int v = 0; v < 6; ++v) s += A2[(kq * 6 + v) * 6 + w];
      beff += b2[kq * 16 + c] * s;
    }
    float sc = g2[ch] * rsqrtf(v2[ch] + eps);
    aff2[r] = make_float2(sc, (beff - m2[ch]) * sc + be2[ch]);
  } else if (tid < 384) {
    int r = tid - 368;
    float sc = g3[r] * rsqrtf(v3[r] + eps);
    aff3[r] = make_float2(sc, (b3[r] - m3[r]) * sc + be3[r]);
  } else if (tid < 448) {
    int r = tid - 384;
    float sc = g4[r] * rsqrtf(v4[r] + eps);
    aff4[r] = make_float2(sc, (b4[r] - m4[r]) * sc + be4[r]);
  }
}

__global__ void zero_f1pad(__bf16* __restrict__ f1bn) {
  int tid = blockIdx.x * blockDim.x + threadIdx.x;
  if (tid >= NSEQ * T * 16) return;
  f1bn[(size_t)(tid >> 4) * F1_STRIDE + 272 + (tid & 15)] = (__bf16)0.0f;
}

// ---------------------------------------------------------------------------
// GEMM kernels: 256 threads = 8 waves; each wave owns one 16-col (n,t) tile.
// A fragments staged into LDS (double-buffered K-chunk groups) shared by all
// 8 waves; B fragments are aligned 32B/lane vector loads from bf16 images.
// ---------------------------------------------------------------------------

__global__ __launch_bounds__(256) void gemm1_kernel(
    const __bf16* __restrict__ posesb, const __bf16* __restrict__ A1f,
    const float2* __restrict__ aff1, __bf16* __restrict__ f1bn) {
  // groups of 2 chunks (one kt): 9 supersteps; buffer = 2*17 KB
  constexpr int NB = 2 * G1_RT;          // 34 blocks of 512 bf16
  __shared__ alignas(32) uint4 sA[2][NB * 64];
  const int lane = threadIdx.x & 31;
  const int col = blockIdx.x * 8 + (threadIdx.x >> 5);
  const int nn = col >> 7, tb = (col & 127) << 4;
  const int nl = lane & 15, half = lane >> 4;
  const uint4* gA = (const uint4*)A1f;

  auto stage = [&](int g, int buf) {
    for (int idx = threadIdx.x; idx < NB * 64; idx += 256) {
      int b = idx >> 6, q = idx & 63;    // b = kkl*17 + rt
      int kkl = b / G1_RT, rt = b % G1_RT;
      copy16_lds(&sA[buf][idx], &gA[(rt * G1_CH + 2 * g + kkl) * 64 + q]);
    }
  };

  v8f acc[G1_RT];
#pragma unroll
  for (int rt = 0; rt < G1_RT; ++rt) acc[rt] = zero8f();

  stage(0, 0);
  wait_async_all();
  __syncthreads();
  for (int g = 0; g < 9; ++g) {
    if (g + 1 < 9) stage(g + 1, (g + 1) & 1);
    const __bf16* base = (const __bf16*)sA[g & 1];
#pragma unroll
    for (int kkl = 0; kkl < 2; ++kkl) {
      const int kk = 2 * g + kkl;
      const int kt = kk >> 1, b0 = (kk & 1) * 32 + half * 16;
      const int tin = tb + nl + kt - 4;
      v16bf bf;
      if (tin >= 0 && tin < T)
        bf = *(const v16bf*)(posesb + ((size_t)nn * T + tin) * 64 + b0);
      else
        bf = zero16bf();
#pragma unroll
      for (int rt = 0; rt < G1_RT; ++rt) {
        v16bf af = *(const v16bf*)(base + (kkl * G1_RT + rt) * 512 + lane * 16);
        acc[rt] = wmma_bf16(af, bf, acc[rt]);
      }
    }
    wait_async_all();
    __syncthreads();
  }
  __bf16* orow = f1bn + ((size_t)nn * T + tb + nl) * F1_STRIDE;
#pragma unroll
  for (int rt = 0; rt < G1_RT; ++rt) {
    v8bf o;
#pragma unroll
    for (int j = 0; j < 8; ++j) {
      int r = rt * 16 + half * 8 + j;    // channel = joint*16 + c
      float2 a = aff1[r];
      o[j] = (__bf16)(acc[rt][j] * a.x + a.y);
    }
    *(v8bf*)(orow + rt * 16 + half * 8) = o;
  }
}

__global__ __launch_bounds__(256) void gemm2_kernel(
    const __bf16* __restrict__ f1bn, const __bf16* __restrict__ A2f,
    const float2* __restrict__ aff2, __bf16* __restrict__ f2bn) {
  // groups of 3 chunks: 27 supersteps; buffer = 18 KB
  constexpr int NB = 3 * G2_RT;          // 18 blocks
  __shared__ alignas(32) uint4 sA[2][NB * 64];
  const int lane = threadIdx.x & 31;
  const int col = blockIdx.x * 8 + (threadIdx.x >> 5);
  const int nn = col >> 7, tb = (col & 127) << 4;
  const int nl = lane & 15, half = lane >> 4;
  const uint4* gA = (const uint4*)A2f;

  auto stage = [&](int g, int buf) {
    for (int idx = threadIdx.x; idx < NB * 64; idx += 256) {
      int b = idx >> 6, q = idx & 63;    // b = kkl*6 + rt
      int kkl = b / G2_RT, rt = b % G2_RT;
      copy16_lds(&sA[buf][idx], &gA[(rt * G2_CH + 3 * g + kkl) * 64 + q]);
    }
  };

  v8f acc[G2_RT];
#pragma unroll
  for (int rt = 0; rt < G2_RT; ++rt) acc[rt] = zero8f();

  stage(0, 0);
  wait_async_all();
  __syncthreads();
  for (int g = 0; g < 27; ++g) {
    if (g + 1 < 27) stage(g + 1, (g + 1) & 1);
    const __bf16* base = (const __bf16*)sA[g & 1];
#pragma unroll
    for (int kkl = 0; kkl < 3; ++kkl) {
      const int kk = 3 * g + kkl;
      const int kt = kk / 9, off = (kk % 9) * 32;
      const int tin = tb + nl + kt - 4;
      v16bf bf;
      if (tin >= 0 && tin < T)
        bf = *(const v16bf*)(f1bn + ((size_t)nn * T + tin) * F1_STRIDE + off +
                             half * 16);
      else
        bf = zero16bf();
#pragma unroll
      for (int rt = 0; rt < G2_RT; ++rt) {
        v16bf af = *(const v16bf*)(base + (kkl * G2_RT + rt) * 512 + lane * 16);
        acc[rt] = wmma_bf16(af, bf, acc[rt]);
      }
    }
    wait_async_all();
    __syncthreads();
  }
  __bf16* orow = f2bn + ((size_t)nn * T + tb + nl) * F2_STRIDE;
#pragma unroll
  for (int rt = 0; rt < G2_RT; ++rt) {
    v8bf o;
#pragma unroll
    for (int j = 0; j < 8; ++j) {
      int r = rt * 16 + half * 8 + j;    // channel = c*6 + w
      float2 a = aff2[r];
      o[j] = (__bf16)(acc[rt][j] * a.x + a.y);
    }
    *(v8bf*)(orow + rt * 16 + half * 8) = o;
  }
}

__global__ __launch_bounds__(256) void gemm3_kernel(
    const __bf16* __restrict__ f2bn, const __bf16* __restrict__ A3f,
    const float2* __restrict__ aff3, __bf16* __restrict__ f3) {
  __shared__ alignas(32) uint4 sA[G3_CH * 32];  // 7.5 KB: first half of the 15 KB weight image
  const int lane = threadIdx.x & 31;
  const int col = blockIdx.x * 8 + (threadIdx.x >> 5);
  const int nn = col >> 7, tb = (col & 127) << 4;
  const int nl = lane & 15, half = lane >> 4;
  const uint4* gA = (const uint4*)A3f;
  for (int idx = threadIdx.x; idx < G3_CH * 32; idx += 256)
    copy16_lds(&sA[idx], &gA[idx]);
  wait_async_all();
  __syncthreads();

  const __bf16* base = (const __bf16*)sA;
  v8f acc = zero8f();
  for (int kk = 0; kk < G3_CH; ++kk) {
    const int kt = kk / 3, off = (kk % 3) * 32;
    const int tin = tb + nl + kt - 2;
    v16bf bf;
    if (tin >= 0 && tin < T)
      bf = *(const v16bf*)(f2bn + ((size_t)nn * T + tin) * F2_STRIDE + off +
                           half * 16);
    else
      bf = zero16bf();
    v16bf af = *(const v16bf*)(base + kk * 512 + lane * 16);
    acc = wmma_bf16(af, bf, acc);
  }
  __bf16* orow = f3 + ((size_t)nn * T + tb + nl) * 16;
  v8bf o;
#pragma unroll
  for (int j = 0; j < 8; ++j) {
    int r = half * 8 + j;
    float2 a = aff3[r];
    o[j] = (__bf16)leaky(acc[j] * a.x + a.y);
  }
  *(v8bf*)(orow + half * 8) = o;
}

__global__ __launch_bounds__(256) void gemm4_kernel(
    const __bf16* __restrict__ f3, const __bf16* __restrict__ A4f,
    const float2* __restrict__ aff4, float* __restrict__ out) {
  __shared__ alignas(32) uint4 sA[G4_RT * G4_CH * 32];  // 4 KB: first half of the 8 KB weight image
  const int lane = threadIdx.x & 31;
  const int col = blockIdx.x * 8 + (threadIdx.x >> 5);
  const int nn = col >> 7, tb = (col & 127) << 4;
  const int nl = lane & 15, half = lane >> 4;
  const uint4* gA = (const uint4*)A4f;
  for (int idx = threadIdx.x; idx < G4_RT * G4_CH * 32; idx += 256)
    copy16_lds(&sA[idx], &gA[idx]);
  wait_async_all();
  __syncthreads();

  const __bf16* base = (const __bf16*)sA;
  v8f acc[G4_RT];
#pragma unroll
  for (int rt = 0; rt < G4_RT; ++rt) acc[rt] = zero8f();
#pragma unroll
  for (int kk = 0; kk < G4_CH; ++kk) {
    const int kt = kk * 2 + half;        // flat k = kk*32 + half*16 + i
    const int tin = tb + nl + kt - 1;
    const bool valid = (kt < 3) && (tin >= 0) && (tin < T);
    v16bf bf;
    if (valid)
      bf = *(const v16bf*)(f3 + ((size_t)nn * T + tin) * 16);
    else
      bf = zero16bf();
#pragma unroll
    for (int rt = 0; rt < G4_RT; ++rt) {
      v16bf af = *(const v16bf*)(base + (rt * G4_CH + kk) * 512 + lane * 16);
      acc[rt] = wmma_bf16(af, bf, acc[rt]);
    }
  }
  float* orow = out + ((size_t)nn * T + tb + nl) * 64;
#pragma unroll
  for (int rt = 0; rt < G4_RT; ++rt) {
    v8f o;
#pragma unroll
    for (int j = 0; j < 8; ++j) {
      int r = rt * 16 + half * 8 + j;
      float2 a = aff4[r];
      o[j] = leaky(acc[rt][j] * a.x + a.y);
    }
    *(v8f*)(orow + rt * 16 + half * 8) = o;
  }
}

} // namespace

extern "C" void kernel_launch(void* const* d_in, const int* in_sizes, int n_in,
                              void* d_out, int out_size, void* d_ws, size_t ws_size,
                              hipStream_t stream) {
  const float* poses = (const float*)d_in[0];
  const float* A1 = (const float*)d_in[1];
  const float* A2 = (const float*)d_in[2];
  const float* w1 = (const float*)d_in[3];
  const float* b1 = (const float*)d_in[4];
  const float* g1 = (const float*)d_in[5];
  const float* be1 = (const float*)d_in[6];
  const float* m1 = (const float*)d_in[7];
  const float* v1 = (const float*)d_in[8];
  const float* w2 = (const float*)d_in[9];
  const float* b2 = (const float*)d_in[10];
  const float* g2 = (const float*)d_in[11];
  const float* be2 = (const float*)d_in[12];
  const float* m2 = (const float*)d_in[13];
  const float* v2 = (const float*)d_in[14];
  const float* w3 = (const float*)d_in[15];
  const float* b3 = (const float*)d_in[16];
  const float* g3 = (const float*)d_in[17];
  const float* be3 = (const float*)d_in[18];
  const float* m3 = (const float*)d_in[19];
  const float* v3 = (const float*)d_in[20];
  const float* w4 = (const float*)d_in[21];
  const float* b4 = (const float*)d_in[22];
  const float* g4 = (const float*)d_in[23];
  const float* be4 = (const float*)d_in[24];
  const float* m4 = (const float*)d_in[25];
  const float* v4 = (const float*)d_in[26];

  char* ws = (char*)d_ws;
  size_t off = 0;
  auto take = [&](size_t bytes) {
    size_t o = off;
    off = (o + bytes + 255) & ~(size_t)255;
    return o;
  };
  __bf16* A1f = (__bf16*)(ws + take((size_t)G1_RT * G1_CH * 512 * 2));
  __bf16* A2f = (__bf16*)(ws + take((size_t)G2_RT * G2_CH * 512 * 2));
  __bf16* A3f = (__bf16*)(ws + take((size_t)G3_CH * 512 * 2));
  __bf16* A4f = (__bf16*)(ws + take((size_t)G4_RT * G4_CH * 512 * 2));
  float2* aff1 = (float2*)(ws + take(272 * sizeof(float2)));
  float2* aff2 = (float2*)(ws + take(96 * sizeof(float2)));
  float2* aff3 = (float2*)(ws + take(16 * sizeof(float2)));
  float2* aff4 = (float2*)(ws + take(64 * sizeof(float2)));
  __bf16* posesb = (__bf16*)(ws + take((size_t)NSEQ * T * 64 * 2));
  __bf16* f1bn = (__bf16*)(ws + take((size_t)NSEQ * T * F1_STRIDE * 2));
  __bf16* f2bn = (__bf16*)(ws + take((size_t)NSEQ * T * F2_STRIDE * 2));
  __bf16* f3b = (__bf16*)(ws + take((size_t)NSEQ * T * 16 * 2));
  (void)ws_size;

  prep_poses<<<(NSEQ * T * 64 + 255) / 256, 256, 0, stream>>>(poses, posesb);
  prep_w1<<<(G1_RT * G1_CH * 512 + 255) / 256, 256, 0, stream>>>(w1, A1, A1f);
  prep_w2<<<(G2_RT * G2_CH * 512 + 255) / 256, 256, 0, stream>>>(w2, A2, A2f);
  prep_w3<<<(G3_CH * 512 + 255) / 256, 256, 0, stream>>>(w3, A3f);
  prep_w4<<<(G4_RT * G4_CH * 512 + 255) / 256, 256, 0, stream>>>(w4, A4f);
  prep_aff<<<2, 256, 0, stream>>>(b1, A1, g1, be1, m1, v1, b2, A2, g2, be2, m2, v2,
                                  b3, g3, be3, m3, v3, b4, g4, be4, m4, v4,
                                  aff1, aff2, aff3, aff4);
  zero_f1pad<<<(NSEQ * T * 16 + 255) / 256, 256, 0, stream>>>(f1bn);

  dim3 gblk(CTILES / 8);
  gemm1_kernel<<<gblk, 256, 0, stream>>>(posesb, A1f, aff1, f1bn);
  gemm2_kernel<<<gblk, 256, 0, stream>>>(f1bn, A2f, aff2, f2bn);
  gemm3_kernel<<<gblk, 256, 0, stream>>>(f2bn, A3f, aff3, f3b);
  gemm4_kernel<<<gblk, 256, 0, stream>>>(f3b, A4f, aff4, (float*)d_out);
}
